// PathFeature_33114197852441
// MI455X (gfx1250) — matrix operand order repaired and checked by reading, not compiled, not run on hardware
//
#include <hip/hip_runtime.h>
#include <hip/hip_bf16.h>

typedef __attribute__((ext_vector_type(16))) __bf16 v16bf;
typedef __attribute__((ext_vector_type(8)))  float  v8f;

#define D 256

// ---------- helpers ----------
static __device__ __forceinline__ unsigned short f2bf(float f) {
    unsigned u = __float_as_uint(f);
    unsigned r = (u + 0x7FFFu + ((u >> 16) & 1u)) >> 16;  // RNE
    return (unsigned short)r;
}

union FragBF {
    v16bf v;
    uint4 q[2];
};

// A-fragment (16x32 bf16, ISA 7.12.2): per lane two 16B chunks at +0 and +16 elems
static __device__ __forceinline__ v16bf load_fragA(const unsigned short* p) {
    FragBF f;
    f.q[0] = *(const uint4*)(p);
    f.q[1] = *(const uint4*)(p + 16);
    return f.v;
}
// B-fragment: per lane two 16B chunks at +0 and +16 elems (from transposed weight matrix)
static __device__ __forceinline__ v16bf load_fragB(const unsigned short* p) {
    FragBF f;
    f.q[0] = *(const uint4*)(p);
    f.q[1] = *(const uint4*)(p + 16);
    return f.v;
}

// ---------- kernels ----------
__global__ void zero_f32_kernel(float* __restrict__ p, long long n) {
    long long i = (long long)blockIdx.x * blockDim.x + threadIdx.x;
    if (i < n) p[i] = 0.0f;
}

// 64 threads per edge, float4 per thread, atomic adds into L2-resident accumulator
__global__ void scatter_kernel(const float* __restrict__ feat,
                               const long long* __restrict__ ei,
                               float* __restrict__ agg, float* __restrict__ deg,
                               int E) {
    long long t = (long long)blockIdx.x * blockDim.x + threadIdx.x;
    int edge = (int)(t >> 6);
    if (edge >= E) return;
    int sub = (int)(t & 63);
    int src = (int)ei[edge];
    int dst = (int)ei[(long long)E + edge];
    const float4 v = ((const float4*)(feat + (size_t)src * D))[sub];
    float* a = agg + (size_t)dst * D + sub * 4;
    atomicAdd(a + 0, v.x);
    atomicAdd(a + 1, v.y);
    atomicAdd(a + 2, v.z);
    atomicAdd(a + 3, v.w);
    if (deg && sub == 0) atomicAdd(&deg[dst], 1.0f);
}

__global__ void convert_h_kernel(const float* __restrict__ h,
                                 unsigned short* __restrict__ hbf, long long n) {
    long long i = (long long)blockIdx.x * blockDim.x + threadIdx.x;
    if (i < n) hbf[i] = f2bf(h[i]);
}

__global__ void convert_mean_kernel(const float* __restrict__ agg,
                                    const float* __restrict__ deg,
                                    unsigned short* __restrict__ mbf, long long n) {
    long long i = (long long)blockIdx.x * blockDim.x + threadIdx.x;
    if (i >= n) return;
    int node = (int)(i >> 8);
    mbf[i] = f2bf(agg[i] / fmaxf(deg[node], 1.0f));
}

// WlT[n*D+k] = bf16(Wl[k*D+n]) (transpose so B-fragment loads are contiguous)
__global__ void convert_wT_kernel(const float* __restrict__ Wl,
                                  const float* __restrict__ Wr,
                                  unsigned short* __restrict__ WlT,
                                  unsigned short* __restrict__ WrT) {
    int i = blockIdx.x * blockDim.x + threadIdx.x;
    if (i >= D * D) return;
    int n = i >> 8, kk = i & 255;
    WlT[i] = f2bf(Wl[kk * D + n]);
    WrT[i] = f2bf(Wr[kk * D + n]);
}

// X = relu(mean @ Wl + h @ Wr + bl) via bf16 WMMA, fp32 accumulate.
// One wave per 16x16 output tile; 16 tiles across the 256-wide N dim.
__global__ void gemm_wmma_kernel(const unsigned short* __restrict__ Mbf,
                                 const unsigned short* __restrict__ Hbf,
                                 const unsigned short* __restrict__ WlT,
                                 const unsigned short* __restrict__ WrT,
                                 const float* __restrict__ bl,
                                 float* __restrict__ X, int N) {
    int gwave = (blockIdx.x * blockDim.x + threadIdx.x) >> 5;
    int lane  = threadIdx.x & 31;
    int tileM = gwave >> 4;
    int tileN = gwave & 15;
    if (tileM * 16 >= N) return;                 // wave-uniform exit (EXEC stays all-ones)

    int l15 = lane & 15;
    int hi  = lane >> 4;
    const unsigned short* aRow = Mbf + (size_t)(tileM * 16 + l15) * D + hi * 8;
    const unsigned short* hRow = Hbf + (size_t)(tileM * 16 + l15) * D + hi * 8;
    const unsigned short* bL   = WlT + (size_t)(tileN * 16 + l15) * D + hi * 8;
    const unsigned short* bR   = WrT + (size_t)(tileN * 16 + l15) * D + hi * 8;

    v8f acc = {0.f, 0.f, 0.f, 0.f, 0.f, 0.f, 0.f, 0.f};
#pragma unroll
    for (int k0 = 0; k0 < D; k0 += 32) {
        __builtin_prefetch(aRow + k0 + 64, 0, 0);
        v16bf a0 = load_fragA(aRow + k0);
        v16bf b0 = load_fragB(bL + k0);
        acc = __builtin_amdgcn_wmma_f32_16x16x32_bf16(false, a0, false, b0,
                                                      (short)0, acc, false, false);
        v16bf a1 = load_fragA(hRow + k0);
        v16bf b1 = load_fragB(bR + k0);
        acc = __builtin_amdgcn_wmma_f32_16x16x32_bf16(false, a1, false, b1,
                                                      (short)0, acc, false, false);
    }
    // C/D layout: VGPR r -> lanes0-15 M=r, lanes16-31 M=r+8; col = lane&15
    int orow = hi * 8;
    float bias = bl[tileN * 16 + l15];
#pragma unroll
    for (int rr = 0; rr < 8; ++rr) {
        float v = acc[rr] + bias;
        X[(size_t)(tileM * 16 + rr + orow) * D + tileN * 16 + l15] = v > 0.f ? v : 0.f;
    }
}

// score[i] = tanh(s_agg[i].Wrel + b_rel + x[i].Wroot); also sortable key
__global__ void score_kernel(const float* __restrict__ x,
                             const float* __restrict__ sagg,
                             const float* __restrict__ Wrel,
                             const float* __restrict__ brel,
                             const float* __restrict__ Wroot,
                             float* __restrict__ score,
                             unsigned* __restrict__ key, int N) {
    int gwave = (blockIdx.x * blockDim.x + threadIdx.x) >> 5;
    int lane  = threadIdx.x & 31;
    if (gwave >= N) return;
    const float* xr = x + (size_t)gwave * D + lane * 8;
    const float* sr = sagg + (size_t)gwave * D + lane * 8;
    float s = 0.f;
#pragma unroll
    for (int j = 0; j < 8; ++j)
        s += sr[j] * Wrel[lane * 8 + j] + xr[j] * Wroot[lane * 8 + j];
    for (int off = 16; off; off >>= 1) s += __shfl_xor(s, off);
    if (lane == 0) {
        float sc = tanhf(s + brel[0]);
        score[gwave] = sc;
        unsigned u = __float_as_uint(sc);
        key[gwave] = (u & 0x80000000u) ? ~u : (u | 0x80000000u);
    }
}

// single block: 32-step binary search for k-th largest key; also resets accumulators
__global__ void topk_kernel(const unsigned* __restrict__ key, int N, int k,
                            unsigned* __restrict__ misc, float* __restrict__ num) {
    __shared__ unsigned s_lo, s_hi, s_cnt;
    int tid = threadIdx.x;
    if (tid == 0) { s_lo = 0u; s_hi = 0xFFFFFFFFu; }
    __syncthreads();
    for (int it = 0; it < 33; ++it) {
        unsigned lo = s_lo, hi = s_hi;
        if (lo >= hi) break;
        unsigned mid = lo + ((hi - lo) >> 1) + ((hi - lo) & 1u);
        if (tid == 0) s_cnt = 0u;
        __syncthreads();
        unsigned c = 0;
        for (int i = tid; i < N; i += 1024) c += (key[i] >= mid) ? 1u : 0u;
        atomicAdd(&s_cnt, c);
        __syncthreads();
        if (tid == 0) {
            if (s_cnt >= (unsigned)k) s_lo = mid; else s_hi = mid - 1u;
        }
        __syncthreads();
    }
    unsigned T = s_lo;
    if (tid == 0) s_cnt = 0u;
    __syncthreads();
    unsigned c = 0;
    for (int i = tid; i < N; i += 1024) c += (key[i] > T) ? 1u : 0u;
    atomicAdd(&s_cnt, c);
    __syncthreads();
    if (tid == 0) {
        misc[0] = T;              // k-th key
        misc[1] = s_cnt;          // count strictly greater
        misc[2] = 0u;             // tie slot counter
        ((float*)misc)[4] = 0.f;  // sumexp accumulator
    }
    if (tid < 256) num[tid] = 0.f;
}

// gate logit for selected nodes; -inf for rejected
__global__ void selgate_kernel(const float* __restrict__ x,
                               const float* __restrict__ score,
                               const unsigned* __restrict__ key,
                               const float* __restrict__ Wgate,
                               const float* __restrict__ bgate,
                               unsigned* __restrict__ misc,
                               float* __restrict__ g, int N, int k) {
    int gwave = (blockIdx.x * blockDim.x + threadIdx.x) >> 5;
    int lane  = threadIdx.x & 31;
    if (gwave >= N) return;
    unsigned T = misc[0], cg = misc[1];
    unsigned ky = key[gwave];
    int sel = 0;
    if (ky > T) {
        sel = 1;
    } else if (ky == T) {
        int pos = 0;
        if (lane == 0) pos = (int)atomicAdd(&misc[2], 1u);
        pos = __shfl(pos, 0);
        sel = ((unsigned)pos < (unsigned)k - cg) ? 1 : 0;
    }
    if (!sel) {
        if (lane == 0) g[gwave] = -__builtin_inff();
        return;
    }
    const float* xr = x + (size_t)gwave * D + lane * 8;
    float s = 0.f;
#pragma unroll
    for (int j = 0; j < 8; ++j) s += xr[j] * Wgate[lane * 8 + j];
    for (int off = 16; off; off >>= 1) s += __shfl_xor(s, off);
    if (lane == 0) g[gwave] = score[gwave] * s + bgate[0];
}

__global__ void max_kernel(const float* __restrict__ g, int N,
                           unsigned* __restrict__ misc) {
    __shared__ float sm[1024];
    int tid = threadIdx.x;
    float v = -__builtin_inff();
    for (int i = tid; i < N; i += 1024) v = fmaxf(v, g[i]);
    sm[tid] = v;
    __syncthreads();
    for (int s = 512; s; s >>= 1) {
        if (tid < s) sm[tid] = fmaxf(sm[tid], sm[tid + s]);
        __syncthreads();
    }
    if (tid == 0) ((float*)misc)[3] = sm[0];
}

// 32 nodes per 256-thread block: partial weighted sums in LDS, then global atomics
__global__ void accum_kernel(const float* __restrict__ x,
                             const float* __restrict__ score,
                             const float* __restrict__ g,
                             const unsigned* __restrict__ misc,
                             float* __restrict__ num, int N) {
    __shared__ float snum[256];
    __shared__ float ssum;
    int tid = threadIdx.x;
    int base = blockIdx.x * 32;
    float m = ((const float*)misc)[3];
    snum[tid] = 0.f;
    if (tid == 0) ssum = 0.f;
    __syncthreads();
    for (int j = 0; j < 32; ++j) {
        int i = base + j;
        if (i >= N) break;
        float gi = g[i];
        if (gi > -1.0e37f) {
            float wgt = expf(gi - m);
            snum[tid] += wgt * score[i] * x[(size_t)i * D + tid];
            if (tid == 0) ssum += wgt;
        }
    }
    __syncthreads();
    atomicAdd(&num[tid], snum[tid]);
    if (tid == 0) atomicAdd((float*)&((const unsigned*)misc)[4] - 0, ssum);
}

__global__ void finalize_kernel(const float* __restrict__ num,
                                const unsigned* __restrict__ misc,
                                float* __restrict__ out) {
    int tid = threadIdx.x;
    out[tid] = num[tid] / ((const float*)misc)[4];
}

// ---------- launcher ----------
extern "C" void kernel_launch(void* const* d_in, const int* in_sizes, int n_in,
                              void* d_out, int out_size, void* d_ws, size_t ws_size,
                              hipStream_t stream) {
    const int N = in_sizes[0] / D;
    const int E = in_sizes[1] / 2;
    const int k = (int)((4LL * (long long)N + 4) / 5);  // ceil(0.8*N)

    const float*     h     = (const float*)d_in[0];
    const long long* ei    = (const long long*)d_in[1];
    const float*     Wl    = (const float*)d_in[2];
    const float*     bl    = (const float*)d_in[3];
    const float*     Wr    = (const float*)d_in[4];
    const float*     Wrel  = (const float*)d_in[5];
    const float*     brel  = (const float*)d_in[6];
    const float*     Wroot = (const float*)d_in[7];
    const float*     Wgate = (const float*)d_in[8];
    const float*     bgate = (const float*)d_in[9];
    float* out = (float*)d_out;
    (void)n_in; (void)out_size; (void)ws_size;

    char* w = (char*)d_ws;
    size_t off = 0;
    auto alloc = [&](size_t bytes) -> void* {
        off = (off + 255) & ~(size_t)255;
        void* p = w + off;
        off += bytes;
        return p;
    };
    long long nd = (long long)N * D;
    float*          agg  = (float*)alloc((size_t)nd * 4);   // reused as s_agg
    float*          deg  = (float*)alloc((size_t)N * 4);
    float*          x    = (float*)alloc((size_t)nd * 4);
    float*          scr  = (float*)alloc((size_t)N * 4);
    unsigned*       key  = (unsigned*)alloc((size_t)N * 4);
    float*          g    = (float*)alloc((size_t)N * 4);
    unsigned short* hbf  = (unsigned short*)alloc((size_t)nd * 2);
    unsigned short* mbf  = (unsigned short*)alloc((size_t)nd * 2);
    unsigned short* WlT  = (unsigned short*)alloc((size_t)D * D * 2);
    unsigned short* WrT  = (unsigned short*)alloc((size_t)D * D * 2);
    unsigned*       misc = (unsigned*)alloc(64 * 4);
    float*          num  = (float*)alloc(256 * 4);

    int bndN  = (int)((nd + 255) / 256);
    int bN    = (N + 255) / 256;
    long long eThreads = (long long)E * 64;
    int bE    = (int)((eThreads + 255) / 256);
    int bWave = (N * 32 + 255) / 256;  // wave-per-node kernels

    // 1) SAGEConv mean aggregation (L2-resident atomics)
    zero_f32_kernel<<<bndN, 256, 0, stream>>>(agg, nd);
    zero_f32_kernel<<<bN, 256, 0, stream>>>(deg, N);
    scatter_kernel<<<bE, 256, 0, stream>>>(h, ei, agg, deg, E);

    // 2) bf16 conversions (features + transposed weights)
    convert_h_kernel<<<bndN, 256, 0, stream>>>(h, hbf, nd);
    convert_mean_kernel<<<bndN, 256, 0, stream>>>(agg, deg, mbf, nd);
    convert_wT_kernel<<<(D * D + 255) / 256, 256, 0, stream>>>(Wl, Wr, WlT, WrT);

    // 3) fused dual GEMM + bias + ReLU via v_wmma_f32_16x16x32_bf16
    int tilesM = (N + 15) / 16;
    int waves  = tilesM * 16;
    gemm_wmma_kernel<<<(waves * 32 + 255) / 256, 256, 0, stream>>>(
        mbf, hbf, WlT, WrT, bl, x, N);

    // 4) GraphConv add-aggregation on x, then score/tanh/key
    zero_f32_kernel<<<bndN, 256, 0, stream>>>(agg, nd);
    scatter_kernel<<<bE, 256, 0, stream>>>(x, ei, agg, nullptr, E);
    score_kernel<<<bWave, 256, 0, stream>>>(x, agg, Wrel, brel, Wroot, scr, key, N);

    // 5) top-k threshold (binary search on sortable keys), selection + gated readout
    topk_kernel<<<1, 1024, 0, stream>>>(key, N, k, misc, num);
    selgate_kernel<<<bWave, 256, 0, stream>>>(x, scr, key, Wgate, bgate, misc, g, N, k);
    max_kernel<<<1, 1024, 0, stream>>>(g, N, misc);
    accum_kernel<<<(N + 31) / 32, 256, 0, stream>>>(x, scr, g, misc, num, N);
    finalize_kernel<<<1, 256, 0, stream>>>(num, misc, out);
}
